// CPC_80736795230858
// MI455X (gfx1250) — compile-verified, matrix operand order checked by reading, not run on hardware
//
#include <hip/hip_runtime.h>
#include <hip/hip_bf16.h>
#include <math.h>

typedef __bf16 bf16_t;
typedef __attribute__((ext_vector_type(16))) __bf16 v16bf;
typedef __attribute__((ext_vector_type(8)))  float  v8f;
typedef __attribute__((ext_vector_type(2)))  __bf16 bf16x2;

#define SEQ_LEN 8192
#define DIM     256
#define HID     256
#define G3      768     // 3*HID
#define TSTART  1024    // SEQ_LEN//8
#define TEND    6815    // SEQ_LEN - 10 - 1365 - 4 + 2
#define TCNT    5791    // TEND - TSTART
#define NEGD    1365    // SEQ_LEN//6
#define NSAMP   10
#define TSPAN   4
#define EPS_F   1e-8f

// ---- workspace layout (bytes) ----
static constexpr size_t OFF_DATA_BF = 0;                                    // 8192*256 bf16
static constexpr size_t OFF_WIH_BF  = OFF_DATA_BF + (size_t)SEQ_LEN*DIM*2;  // 768*256 bf16
static constexpr size_t OFF_WHHT_BF = OFF_WIH_BF  + (size_t)G3*DIM*2;       // 256*768 bf16 (transposed)
static constexpr size_t OFF_XW      = OFF_WHHT_BF + (size_t)DIM*G3*2;       // 8192*768 f32
static constexpr size_t OFF_ACC     = OFF_XW      + (size_t)SEQ_LEN*G3*4;   // 2 f32

// ---------------- init: zero scoring accumulators ----------------
__global__ void init_kernel(float* accum) {
    if (threadIdx.x < 2) accum[threadIdx.x] = 0.0f;
}

// ---------------- fp32 -> bf16 conversions (+ Whh transpose) ----------------
__global__ void cvt_kernel(const float* __restrict__ data,
                           const float* __restrict__ Wih,
                           const float* __restrict__ Whh,
                           bf16_t* __restrict__ data_bf,
                           bf16_t* __restrict__ wih_bf,
                           bf16_t* __restrict__ whht_bf) {
    const int NDATA = SEQ_LEN * DIM;
    const int NW    = G3 * DIM;
    int i = blockIdx.x * blockDim.x + threadIdx.x;
    if (i < NDATA) {
        data_bf[i] = (bf16_t)data[i];
    } else if (i < NDATA + NW) {
        int e = i - NDATA;
        wih_bf[e] = (bf16_t)Wih[e];
    } else if (i < NDATA + 2 * NW) {
        int e = i - NDATA - NW;        // e = k*G3 + j
        int k = e / G3, j = e % G3;
        whht_bf[e] = (bf16_t)Whh[j * DIM + k];
    }
}

// ---------------- xW = data @ Wih^T + bih  via bf16 WMMA ----------------
// One 16x16 output tile per wave; K=256 as 8 chunks of 32.
__global__ void __launch_bounds__(128)
xw_gemm_kernel(const bf16_t* __restrict__ data_bf,   // [8192][256]
               const bf16_t* __restrict__ wih_bf,    // [768][256]
               const float*  __restrict__ bih,       // [768]
               float* __restrict__ xw)                // [8192][768]
{
    const int wave = threadIdx.x >> 5;
    const int lane = threadIdx.x & 31;
    const int half = lane >> 4;
    const int l16  = lane & 15;

    const int tiles_n = G3 / 16;                 // 48
    const int tile = blockIdx.x * 4 + wave;      // 24576 tiles total
    const int tm = tile / tiles_n;               // 0..511  (M tile)
    const int tn = tile % tiles_n;               // 0..47   (N tile)

    const bf16_t* arow_p = data_bf + (size_t)(tm * 16 + l16) * DIM; // A row (M = l16)
    const bf16_t* bcol_p = wih_bf  + (size_t)(tn * 16 + l16) * DIM; // B col (N = l16) = Wih row

    v8f acc = {};
    #pragma unroll
    for (int kk = 0; kk < DIM; kk += 32) {
        v16bf a, b;
        #pragma unroll
        for (int v = 0; v < 8; ++v) {
            const int k0 = ((v < 4) ? (2 * v) : (16 + 2 * (v - 4))) + 8 * half + kk;
            bf16x2 pa = *(const bf16x2*)(arow_p + k0);
            bf16x2 pb = *(const bf16x2*)(bcol_p + k0);
            a[2 * v] = pa[0]; a[2 * v + 1] = pa[1];
            b[2 * v] = pb[0]; b[2 * v + 1] = pb[1];
        }
        acc = __builtin_amdgcn_wmma_f32_16x16x32_bf16(
            false, a, false, b, (short)0, acc, false, false);
    }

    #pragma unroll
    for (int v = 0; v < 8; ++v) {
        const int m = tm * 16 + v + 8 * half;    // C/D: VGPR v -> M row
        const int n = tn * 16 + l16;             // C/D: lane   -> N col
        xw[(size_t)m * G3 + n] = acc[v] + bih[n];
    }
}

// ---------------- sequential GRU: one persistent workgroup ----------------
// 768 threads: thread j owns gate pre-activation gh[j]; h lives in LDS.
__global__ void __launch_bounds__(768)
gru_kernel(const float*  __restrict__ xw,       // [8192][768] precomputed
           const bf16_t* __restrict__ whht_bf,  // [256][768] transposed bf16
           const float*  __restrict__ bhh,      // [768]
           float* __restrict__ z)               // [8192][256] output hidden states
{
    __shared__ float h_s[HID];
    __shared__ float gh_s[G3];
    const int j = threadIdx.x;
    const float bj = bhh[j];
    if (j < HID) h_s[j] = 0.0f;
    __syncthreads();

    for (int t = 0; t < SEQ_LEN; ++t) {
        // gh[j] = dot(Whh[j,:], h) + bhh[j]; coalesced bf16 weight stream.
        float acc = bj;
        #pragma unroll 8
        for (int k = 0; k < HID; ++k)
            acc += (float)whht_bf[k * G3 + j] * h_s[k];
        gh_s[j] = acc;
        __syncthreads();

        if (j < HID) {
            const float* xr = xw + (size_t)t * G3;
            float r  = 1.0f / (1.0f + __expf(-(xr[j]           + gh_s[j])));
            float zg = 1.0f / (1.0f + __expf(-(xr[HID + j]     + gh_s[HID + j])));
            float n  = tanhf(xr[2 * HID + j] + r * gh_s[2 * HID + j]);
            float hn = (1.0f - zg) * n + zg * h_s[j];
            h_s[j] = hn;
            z[(size_t)t * HID + j] = hn;
        }
        __syncthreads();
    }
}

// ---------------- NCE scoring: cosine sims + log-softmax ----------------
__global__ void __launch_bounds__(256)
score_kernel(const float* __restrict__ x,     // data[0]: [8192][256]
             const float* __restrict__ z,     // [8192][256]
             float* __restrict__ accum)       // [0]=sum(-logp0), [1]=correct
{
    __shared__ float c_s[HID];
    __shared__ float sims[TSPAN][NSAMP];
    const int t    = TSTART + blockIdx.x;
    const int tid  = threadIdx.x;
    const int wave = tid >> 5, lane = tid & 31;

    if (tid < HID) c_s[tid] = z[(size_t)t * HID + tid];
    __syncthreads();

    // context norm (computed per wave; 32-lane shuffle reduce)
    float cs = 0.0f;
    for (int e = lane; e < HID; e += 32) { float v = c_s[e]; cs += v * v; }
    #pragma unroll
    for (int off = 16; off; off >>= 1) cs += __shfl_down(cs, off, 32);
    cs = __shfl(cs, 0, 32);
    const float cn = fmaxf(sqrtf(cs), EPS_F);

    // 40 (s,n) pairs over 8 waves, 5 per wave
    #pragma unroll
    for (int q = 0; q < 5; ++q) {
        const int p = wave * 5 + q;            // 0..39
        const int s = p / NSAMP;               // 0..3   (i = s+1)
        const int n = p % NSAMP;               // 0..9
        const int row = (n == 0) ? (t + s + 1)
                                 : (t + s + 1 + NEGD + (n - 1));
        const float* xr = x + (size_t)row * DIM;
        float dot = 0.0f, ss = 0.0f;
        for (int e = lane; e < DIM; e += 32) {
            float xv = xr[e];
            dot += xv * c_s[e];
            ss  += xv * xv;
        }
        #pragma unroll
        for (int off = 16; off; off >>= 1) {
            dot += __shfl_down(dot, off, 32);
            ss  += __shfl_down(ss,  off, 32);
        }
        if (lane == 0) {
            float sn = fmaxf(sqrtf(ss), EPS_F);
            sims[s][n] = dot / (sn * cn);
        }
    }
    __syncthreads();

    if (tid < TSPAN) {
        float m = sims[tid][0];
        #pragma unroll
        for (int n = 1; n < NSAMP; ++n) m = fmaxf(m, sims[tid][n]);
        float se = 0.0f;
        #pragma unroll
        for (int n = 0; n < NSAMP; ++n) se += __expf(sims[tid][n] - m);
        float logp0 = (sims[tid][0] - m) - __logf(se);
        atomicAdd(&accum[0], -logp0);
        bool best = true;
        #pragma unroll
        for (int n = 1; n < NSAMP; ++n)
            if (sims[tid][n] > sims[tid][0]) best = false;
        if (best) atomicAdd(&accum[1], 1.0f);
    }
}

__global__ void finalize_kernel(const float* __restrict__ accum,
                                float* __restrict__ out_tail) {
    const float denom = (float)(TCNT * TSPAN);
    out_tail[0] = accum[0] / denom;   // nce
    out_tail[1] = accum[1] / denom;   // acc
}

extern "C" void kernel_launch(void* const* d_in, const int* in_sizes, int n_in,
                              void* d_out, int out_size, void* d_ws, size_t ws_size,
                              hipStream_t stream) {
    const float* data = (const float*)d_in[0];   // [1,8192,256]
    const float* Wih  = (const float*)d_in[1];   // [768,256]
    const float* Whh  = (const float*)d_in[2];   // [768,256]
    const float* bih  = (const float*)d_in[3];   // [768]
    const float* bhh  = (const float*)d_in[4];   // [768]
    float* out = (float*)d_out;                  // [z(2097152) | nce | acc]

    char* ws = (char*)d_ws;
    bf16_t* data_bf = (bf16_t*)(ws + OFF_DATA_BF);
    bf16_t* wih_bf  = (bf16_t*)(ws + OFF_WIH_BF);
    bf16_t* whht_bf = (bf16_t*)(ws + OFF_WHHT_BF);
    float*  xw      = (float*)(ws + OFF_XW);
    float*  accum   = (float*)(ws + OFF_ACC);

    init_kernel<<<1, 32, 0, stream>>>(accum);

    const int ncvt = SEQ_LEN * DIM + 2 * G3 * DIM;      // 2,490,368
    cvt_kernel<<<(ncvt + 255) / 256, 256, 0, stream>>>(
        data, Wih, Whh, data_bf, wih_bf, whht_bf);

    const int ntiles = (SEQ_LEN / 16) * (G3 / 16);      // 24576
    xw_gemm_kernel<<<ntiles / 4, 128, 0, stream>>>(data_bf, wih_bf, bih, xw);

    gru_kernel<<<1, 768, 0, stream>>>(xw, whht_bf, bhh, out);

    score_kernel<<<TCNT, 256, 0, stream>>>(data, out, accum);

    finalize_kernel<<<1, 1, 0, stream>>>(accum, out + (size_t)SEQ_LEN * HID);
}
